// _Experts_33758442947147
// MI455X (gfx1250) — matrix operand order repaired and checked by reading, not compiled, not run on hardware
//
#include <hip/hip_runtime.h>
#include <hip/hip_bf16.h>

// Problem constants (from reference)
#define E_ 16
#define H_ 2048
#define I_ 768
#define T_ 2048
#define K_ 2
#define P_ (T_ * K_)       // 4096 routed (token, expert) pairs
#define PPAD_ 4352         // per-expert 16-aligned capacity
#define TILES_MAX (PPAD_ / 16)   // 272
#define MID_OFF 35328      // byte offset of mid[] inside workspace (16B aligned)

typedef __attribute__((ext_vector_type(16))) __bf16 bf16x16;
typedef __attribute__((ext_vector_type(8)))  __bf16 bf16x8;
typedef __attribute__((ext_vector_type(4)))  __bf16 bf16x4;
typedef __attribute__((ext_vector_type(8)))  float  f32x8;
typedef __attribute__((ext_vector_type(4)))  float  f32x4;
typedef __attribute__((ext_vector_type(4)))  int    i32x4;

#if __has_builtin(__builtin_amdgcn_global_load_async_to_lds_b128) && \
    __has_builtin(__builtin_amdgcn_s_wait_asynccnt)
#define USE_ASYNC_LDS 1
typedef __attribute__((address_space(1))) i32x4 gi32x4;   // global int4
typedef __attribute__((address_space(3))) i32x4 li32x4;   // LDS int4
#else
#define USE_ASYNC_LDS 0
#endif

// ---------------- helper: find expert owning tile b ----------------
__device__ __forceinline__ int find_expert(const int* __restrict__ cumTiles, int b) {
    int e = 0;
    while (b >= cumTiles[e + 1]) ++e;   // uniform scalar loop, <=16 iters
    return e;
}

// fast sigmoid via hardware-rate v_rcp_f32 (avoid IEEE div expansion)
__device__ __forceinline__ float fast_sigmoid(float g) {
    return __builtin_amdgcn_rcpf(1.0f + __expf(-g));
}

// ---------------- zero kernels ----------------
__global__ __launch_bounds__(256) void k_zero_out(float* __restrict__ out) {
    int i = blockIdx.x * 256 + threadIdx.x;           // grid sized exactly T*H/4
    f32x4 z = {0.f, 0.f, 0.f, 0.f};
    if (i < (T_ * H_) / 4) ((f32x4*)out)[i] = z;
}

__global__ __launch_bounds__(256) void k_zero_ws(int* __restrict__ wsI) {
    int i = blockIdx.x * 256 + threadIdx.x;
    if (i < 128 + 2 * PPAD_) wsI[i] = 0;              // header + bucket_tok + bucket_w
}

// ---------------- routing ----------------
__global__ __launch_bounds__(256) void k_count(const int* __restrict__ idx,
                                               int* __restrict__ counts) {
    int i = blockIdx.x * 256 + threadIdx.x;
    if (i < P_) atomicAdd(&counts[idx[i]], 1);
}

__global__ __launch_bounds__(32) void k_scan(const int* __restrict__ counts,
                                             int* __restrict__ offsets,
                                             int* __restrict__ cursor,
                                             int* __restrict__ cumTiles) {
    if (threadIdx.x == 0) {
        int tot = 0, ct = 0;
        for (int e = 0; e < E_; ++e) {
            offsets[e]  = tot;
            cursor[e]   = tot;
            cumTiles[e] = ct;
            int pad = (counts[e] + 15) & ~15;         // 16-align each expert bucket
            tot += pad;
            ct  += pad >> 4;
        }
        offsets[E_]  = tot;
        cumTiles[E_] = ct;
    }
}

__global__ __launch_bounds__(256) void k_scatter(const int* __restrict__ idx,
                                                 const float* __restrict__ wts,
                                                 int* __restrict__ cursor,
                                                 int* __restrict__ btok,
                                                 float* __restrict__ bw) {
    int i = blockIdx.x * 256 + threadIdx.x;
    if (i < P_) {
        int e = idx[i];
        int p = atomicAdd(&cursor[e], 1);
        btok[p] = i >> 1;            // token id (K_ == 2)
        bw[p]   = wts[i];
    }
}

// ---------------- gate/up GEMM + SwiGLU: mid[p, I] (bf16) ----------------
__global__ __launch_bounds__(256) void k_gateup(const float* __restrict__ x,
                                                const float* __restrict__ gup,
                                                const int* __restrict__ btok,
                                                const int* __restrict__ cumTiles,
                                                const int* __restrict__ offsets,
                                                __bf16* __restrict__ mid) {
    __shared__ __align__(16) __bf16 sx[16 * H_];      // 64 KB token tile (bf16)
    __shared__ int stok[16];

    int b = blockIdx.x;
    if (b >= cumTiles[E_]) return;
    int e = find_expert(cumTiles, b);
    int pbase = offsets[e] + ((b - cumTiles[e]) << 4);

    int tid = threadIdx.x;
    if (tid < 16) stok[tid] = btok[pbase + tid];      // padded rows -> token 0 (zeroed ws)
    __syncthreads();

    // stage 16 x 2048 fp32 -> bf16 into LDS (float4 loads, bf16x4 stores)
    for (int j = tid; j < (16 * H_) / 4; j += 256) {
        int row = j >> 9;                              // H_/4 = 512 groups per row
        int c4  = (j & 511) << 2;
        f32x4 v = *(const f32x4*)(x + (size_t)stok[row] * H_ + c4);
        bf16x4 h;
        h[0] = (__bf16)v[0]; h[1] = (__bf16)v[1]; h[2] = (__bf16)v[2]; h[3] = (__bf16)v[3];
        *(bf16x4*)(&sx[row * H_ + c4]) = h;
    }
    __syncthreads();

    int wave = tid >> 5;
    int lane = tid & 31;
    int n    = lane & 15;
    int half = lane >> 4;
    int col  = blockIdx.y * 128 + wave * 16 + n;      // intermediate channel in [0, I)

    const float* wg = gup + ((size_t)e * (2 * I_) + col) * H_;         // gate row
    const float* wu = gup + ((size_t)e * (2 * I_) + I_ + col) * H_;    // up row
    const __bf16* xr = &sx[n * H_];                   // A row for this lane (m = n)

    f32x8 accg = {}, accu = {};

    for (int kk = 0; kk < H_; kk += 32) {
        // prefetch next K-chunk of the fp32 weight stream (global_prefetch_b8)
        __builtin_prefetch(wg + kk + 32 + half * 16, 0, 0);
        __builtin_prefetch(wu + kk + 32 + half * 16, 0, 0);

        // A fragment: contiguous 16 K-values at half*16 (matches B)
        bf16x8 alo = *(const bf16x8*)(xr + kk + half * 16);
        bf16x8 ahi = *(const bf16x8*)(xr + kk + half * 16 + 8);
        bf16x16 a;
        #pragma unroll
        for (int j = 0; j < 8; ++j) { a[j] = alo[j]; a[j + 8] = ahi[j]; }

        // B fragments: contiguous 16 K-values at half*16, fp32->bf16 in regs
        int kb = kk + half * 16;
        f32x4 g0 = *(const f32x4*)(wg + kb);
        f32x4 g1 = *(const f32x4*)(wg + kb + 4);
        f32x4 g2 = *(const f32x4*)(wg + kb + 8);
        f32x4 g3 = *(const f32x4*)(wg + kb + 12);
        f32x4 u0 = *(const f32x4*)(wu + kb);
        f32x4 u1 = *(const f32x4*)(wu + kb + 4);
        f32x4 u2 = *(const f32x4*)(wu + kb + 8);
        f32x4 u3 = *(const f32x4*)(wu + kb + 12);
        bf16x16 bg, bu;
        #pragma unroll
        for (int j = 0; j < 4; ++j) {
            bg[j]      = (__bf16)g0[j]; bg[j + 4]  = (__bf16)g1[j];
            bg[j + 8]  = (__bf16)g2[j]; bg[j + 12] = (__bf16)g3[j];
            bu[j]      = (__bf16)u0[j]; bu[j + 4]  = (__bf16)u1[j];
            bu[j + 8]  = (__bf16)u2[j]; bu[j + 12] = (__bf16)u3[j];
        }

        accg = __builtin_amdgcn_wmma_f32_16x16x32_bf16(false, a, false, bg,
                                                       (short)0, accg, false, false);
        accu = __builtin_amdgcn_wmma_f32_16x16x32_bf16(false, a, false, bu,
                                                       (short)0, accu, false, false);
    }

    // SwiGLU + store mid (D layout: M = v + half*8, N = lane&15)
    #pragma unroll
    for (int v = 0; v < 8; ++v) {
        int mrow = v + half * 8;
        float g = accg[v], u = accu[v];
        float m = g * fast_sigmoid(g) * u;
        mid[(size_t)(pbase + mrow) * I_ + col] = (__bf16)m;
    }
}

// ---------------- down GEMM + routed combine into out ----------------
__global__ __launch_bounds__(256) void k_down(const __bf16* __restrict__ mid,
                                              const float* __restrict__ dwn,
                                              const int* __restrict__ btok,
                                              const float* __restrict__ bw,
                                              const int* __restrict__ cumTiles,
                                              const int* __restrict__ offsets,
                                              float* __restrict__ out) {
    __shared__ __align__(16) __bf16 sa[16 * I_];      // 24 KB mid tile (bf16)
    __shared__ int   stok[16];
    __shared__ float sw[16];

    int b = blockIdx.x;
    if (b >= cumTiles[E_]) return;
    int e = find_expert(cumTiles, b);
    int pbase = offsets[e] + ((b - cumTiles[e]) << 4);

    int tid = threadIdx.x;
    if (tid < 16) { stok[tid] = btok[pbase + tid]; sw[tid] = bw[pbase + tid]; }

#if USE_ASYNC_LDS
    // mid is already bf16: DMA it straight into LDS (ASYNCcnt path, no VGPR transit)
    for (int j = tid; j < (16 * I_) / 8; j += 256) {  // 16B chunks
        int row = j / (I_ / 8);
        int c8  = (j - row * (I_ / 8)) << 3;
        __bf16* gsrc = const_cast<__bf16*>(mid) + (size_t)(pbase + row) * I_ + c8;
        __builtin_amdgcn_global_load_async_to_lds_b128(
            (gi32x4*)gsrc, (li32x4*)&sa[row * I_ + c8], 0, 0);
    }
    __builtin_amdgcn_s_wait_asynccnt(0);
    __syncthreads();
#else
    for (int j = tid; j < (16 * I_) / 4; j += 256) {
        int row = j / (I_ / 4);
        int c4  = (j - row * (I_ / 4)) << 2;
        *(bf16x4*)(&sa[row * I_ + c4]) =
            *(const bf16x4*)(mid + (size_t)(pbase + row) * I_ + c4);
    }
    __syncthreads();
#endif

    int wave = tid >> 5;
    int lane = tid & 31;
    int n    = lane & 15;
    int half = lane >> 4;
    int hcol = blockIdx.y * 128 + wave * 16 + n;      // hidden channel in [0, H)

    const float*  wr = dwn + ((size_t)e * H_ + hcol) * I_;
    const __bf16* ar = &sa[n * I_];

    f32x8 acc = {};
    for (int kk = 0; kk < I_; kk += 32) {
        __builtin_prefetch(wr + kk + 32 + half * 16, 0, 0);

        bf16x8 alo = *(const bf16x8*)(ar + kk + half * 16);
        bf16x8 ahi = *(const bf16x8*)(ar + kk + half * 16 + 8);
        bf16x16 a;
        #pragma unroll
        for (int j = 0; j < 8; ++j) { a[j] = alo[j]; a[j + 8] = ahi[j]; }

        int kb = kk + half * 16;
        f32x4 w0 = *(const f32x4*)(wr + kb);
        f32x4 w1 = *(const f32x4*)(wr + kb + 4);
        f32x4 w2 = *(const f32x4*)(wr + kb + 8);
        f32x4 w3 = *(const f32x4*)(wr + kb + 12);
        bf16x16 bb;
        #pragma unroll
        for (int j = 0; j < 4; ++j) {
            bb[j]      = (__bf16)w0[j]; bb[j + 4]  = (__bf16)w1[j];
            bb[j + 8]  = (__bf16)w2[j]; bb[j + 12] = (__bf16)w3[j];
        }

        acc = __builtin_amdgcn_wmma_f32_16x16x32_bf16(false, a, false, bb,
                                                      (short)0, acc, false, false);
    }

    #pragma unroll
    for (int v = 0; v < 8; ++v) {
        int mrow = v + half * 8;
        float val = acc[v] * sw[mrow];                // padded rows: weight 0
        atomicAdd(&out[(size_t)stok[mrow] * H_ + hcol], val);
    }
}

// ---------------- launch ----------------
extern "C" void kernel_launch(void* const* d_in, const int* in_sizes, int n_in,
                              void* d_out, int out_size, void* d_ws, size_t ws_size,
                              hipStream_t stream) {
    const float* x   = (const float*)d_in[0];   // hidden_states [T,H]
    const int*   idx = (const int*)  d_in[1];   // top_k_index   [T,K]
    const float* wts = (const float*)d_in[2];   // top_k_weights [T,K]
    const float* gup = (const float*)d_in[3];   // gate_up_proj  [E,2I,H]
    const float* dwn = (const float*)d_in[4];   // down_proj     [E,H,I]
    float* out = (float*)d_out;

    int* wsI      = (int*)d_ws;
    int* counts   = wsI;            // 16
    int* offsets  = wsI + 16;       // 17
    int* cursor   = wsI + 33;       // 16
    int* cumTiles = wsI + 49;       // 17
    int*   btok = wsI + 128;                    // PPAD_
    float* bw   = (float*)(wsI + 128 + PPAD_);  // PPAD_
    __bf16* mid = (__bf16*)((char*)d_ws + MID_OFF);   // PPAD_ x I_

    k_zero_out<<<(T_ * H_) / 4 / 256, 256, 0, stream>>>(out);
    k_zero_ws <<<(128 + 2 * PPAD_ + 255) / 256, 256, 0, stream>>>(wsI);
    k_count   <<<P_ / 256, 256, 0, stream>>>(idx, counts);
    k_scan    <<<1, 32, 0, stream>>>(counts, offsets, cursor, cumTiles);
    k_scatter <<<P_ / 256, 256, 0, stream>>>(idx, wts, cursor, btok, bw);

    dim3 gB(TILES_MAX, I_ / 128);   // (272, 6)
    k_gateup<<<gB, 256, 0, stream>>>(x, gup, btok, cumTiles, offsets, mid);

    dim3 gC(TILES_MAX, H_ / 128);   // (272, 16)
    k_down<<<gC, 256, 0, stream>>>(mid, dwn, btok, bw, cumTiles, offsets, out);
}
